// Attention_79542794322277
// MI455X (gfx1250) — compile-verified, matrix operand order checked by reading, not run on hardware
//
#include <hip/hip_runtime.h>
#include <hip/hip_bf16.h>

typedef __attribute__((ext_vector_type(16))) __bf16 v16bf;
typedef __attribute__((ext_vector_type(8)))  float  v8f;

#define B_ 4
#define N_ 512
#define D_ 768
#define A_ 128

#define TI_ 8     // i-rows per score block (== waves per block)
#define JT_ 64    // j-chunk staged in LDS

// Native gfx1250 v_tanh_f32 (single TRANS op). Prefer the clang builtin;
// otherwise emit the instruction directly (v_nop after TRANS per ISA 7.4
// co-execution hazard rule before the result is consumed).
#if defined(__has_builtin)
#if __has_builtin(__builtin_amdgcn_tanhf)
#define HAVE_NATIVE_TANH 1
#endif
#endif

#ifndef HAVE_NATIVE_TANH
#warning "gfx1250: __builtin_amdgcn_tanhf not declared; using inline-asm v_tanh_f32"
#endif

static __device__ __forceinline__ float fast_tanh(float x) {
#ifdef HAVE_NATIVE_TANH
  return __builtin_amdgcn_tanhf(x);
#else
  float r;
  asm volatile("v_tanh_f32 %0, %1\n\tv_nop" : "=v"(r) : "v"(x));
  return r;
#endif
}

// ---------------------------------------------------------------------------
// out[r][n] = sum_k mat[r][k] * W[k][n] + bias[n]
// One 16x16 output tile per wave, bf16 WMMA with f32 accumulate, K step 32.
// ---------------------------------------------------------------------------
__global__ __launch_bounds__(256) void proj_wmma(
    const float* __restrict__ mat,   // [B*N, D] f32
    const float* __restrict__ W,     // [D, A]   f32
    const float* __restrict__ bias,  // [A]      f32
    float* __restrict__ out)         // [B*N, A] f32
{
  const int R = B_ * N_;
  int wave = (int)((blockIdx.x * blockDim.x + threadIdx.x) >> 5);
  int lane = (int)(threadIdx.x & 31);
  const int tilesN = A_ / 16;
  int tm = wave / tilesN;
  int tn = wave % tilesN;
  if (tm * 16 >= R) return;           // wave-uniform; EXEC all-ones for WMMA

  int m = lane & 15;
  int g = lane >> 4;                  // K-group select (ISA 16-bit A layout)
  const float* arow = mat + (size_t)(tm * 16 + m) * D_;
  int n = tn * 16 + m;

  v8f acc = {};
#pragma unroll 2
  for (int k0 = 0; k0 < D_; k0 += 32) {
    v16bf av, bv;
#pragma unroll
    for (int e = 0; e < 8; ++e) {
      av[e]     = (__bf16)arow[k0 + g * 8 + e];
      av[e + 8] = (__bf16)arow[k0 + 16 + g * 8 + e];
      bv[e]     = (__bf16)W[(size_t)(k0 + g * 8 + e) * A_ + n];
      bv[e + 8] = (__bf16)W[(size_t)(k0 + 16 + g * 8 + e) * A_ + n];
    }
    acc = __builtin_amdgcn_wmma_f32_16x16x32_bf16(
        false, av, false, bv, (short)0, acc, false, false);
  }

  float bb = bias[n];
  int mb = g * 8;
#pragma unroll
  for (int r = 0; r < 8; ++r)
    out[(size_t)(tm * 16 + mb + r) * A_ + n] = acc[r] + bb;
}

// ---------------------------------------------------------------------------
// Fused pairwise additive scores + mask + softmax.
// One block per (b, 8-row i-tile); wave w owns row i=w. w2 streamed through
// LDS in 64-row chunks (coalesced float4 loads, reused by all 8 rows).
// Per lane: j = lane and j = lane+32, sharing the sw1/sv broadcast reads.
// ---------------------------------------------------------------------------
__global__ __launch_bounds__(256) void score_softmax(
    const float* __restrict__ w1,    // [B*N, A]
    const float* __restrict__ w2,    // [B*N, A]
    const float* __restrict__ vw,    // [A]
    const int*   __restrict__ mask,  // [B, N, N]
    __bf16* __restrict__ attn)       // [B, N, N] bf16
{
  __shared__ __align__(16) float sw1[TI_][A_];       // 4 KB
  __shared__ __align__(16) float sw2[JT_][A_ + 4];   // 33 KB (pad: 2-way banks)
  __shared__ __align__(16) float sv[A_];             // 0.5 KB
  __shared__ __align__(16) float sscore[TI_][N_];    // 16 KB

  const int b    = blockIdx.y;
  const int i0   = blockIdx.x * TI_;
  const int tid  = (int)threadIdx.x;                 // 256 threads = 8 waves
  const int wrow = tid >> 5;                         // wave id == row in tile
  const int jl   = tid & 31;
  const float NEG = -3.402823466e38f;                // jnp.finfo(f32).min

  // stage w1 tile (8x128 = 256 float4) and v
  {
    int r = tid >> 5, c4 = (tid & 31) << 2;          // one float4 each
    *(float4*)&sw1[r][c4] =
        *(const float4*)&w1[(size_t)(b * N_ + i0 + r) * A_ + c4];
    if (tid < A_ / 4)
      *(float4*)&sv[tid << 2] = *(const float4*)&vw[tid << 2];
  }

  for (int c = 0; c < N_ / JT_; ++c) {
    __syncthreads();
    // stage w2 chunk: 64 rows x 128 floats = 2048 float4, 8 per thread
#pragma unroll
    for (int t = 0; t < (JT_ * A_ / 4) / 256; ++t) {
      int flat = tid + 256 * t;
      int jr = flat >> 5;                            // A_/4 = 32 float4/row
      int cc = (flat & 31) << 2;
      *(float4*)&sw2[jr][cc] =
          *(const float4*)&w2[(size_t)(b * N_ + c * JT_ + jr) * A_ + cc];
    }
    __syncthreads();

    // each thread: 2 j's (jl, jl+32) of its wave's row; sw1/sv reads shared
    float a0 = 0.0f, a1 = 0.0f;
#pragma unroll 4
    for (int a = 0; a < A_; a += 4) {
      float4 x1 = *(const float4*)&sw1[wrow][a];     // wave broadcast
      float4 vv = *(const float4*)&sv[a];            // wave broadcast
      float4 p  = *(const float4*)&sw2[jl][a];
      float4 q  = *(const float4*)&sw2[jl + 32][a];
      a0 = fmaf(vv.x, fast_tanh(x1.x + p.x), a0);
      a0 = fmaf(vv.y, fast_tanh(x1.y + p.y), a0);
      a0 = fmaf(vv.z, fast_tanh(x1.z + p.z), a0);
      a0 = fmaf(vv.w, fast_tanh(x1.w + p.w), a0);
      a1 = fmaf(vv.x, fast_tanh(x1.x + q.x), a1);
      a1 = fmaf(vv.y, fast_tanh(x1.y + q.y), a1);
      a1 = fmaf(vv.z, fast_tanh(x1.z + q.z), a1);
      a1 = fmaf(vv.w, fast_tanh(x1.w + q.w), a1);
    }
    sscore[wrow][c * JT_ + jl]      = a0;
    sscore[wrow][c * JT_ + jl + 32] = a1;
  }
  __syncthreads();

  // softmax: one wave per i-row, lanes stride the 512 columns
  {
    size_t gi = (size_t)(b * N_ + i0 + wrow);
    const int* mrow = mask + gi * N_;

    float sval[N_ / 32];
    float mx = NEG;
#pragma unroll
    for (int t = 0; t < N_ / 32; ++t) {
      int j = jl + 32 * t;                           // coalesced mask reads
      float s = mrow[j] ? sscore[wrow][j] : NEG;
      sval[t] = s;
      mx = fmaxf(mx, s);
    }
#pragma unroll
    for (int off = 16; off > 0; off >>= 1)
      mx = fmaxf(mx, __shfl_xor(mx, off, 32));

    float sum = 0.0f;
#pragma unroll
    for (int t = 0; t < N_ / 32; ++t) {
      float e = __builtin_amdgcn_exp2f((sval[t] - mx) * 1.44269504088896f);
      sval[t] = e;
      sum += e;
    }
#pragma unroll
    for (int off = 16; off > 0; off >>= 1)
      sum += __shfl_xor(sum, off, 32);

    float inv = __builtin_amdgcn_rcpf(sum);
    __bf16* arow = attn + gi * N_;
#pragma unroll
    for (int t = 0; t < N_ / 32; ++t)
      arow[jl + 32 * t] = (__bf16)(sval[t] * inv);
  }
}

// ---------------------------------------------------------------------------
// out[b][i][d] = sum_j attn[b][i][j] * mat[b][j][d]   (bf16 WMMA, f32 accum)
// ---------------------------------------------------------------------------
__global__ __launch_bounds__(256) void av_wmma(
    const __bf16* __restrict__ attn, // [B, N, N] bf16
    const float*  __restrict__ mat,  // [B, N, D] f32
    float* __restrict__ out)         // [B, N, D] f32
{
  int wave = (int)((blockIdx.x * blockDim.x + threadIdx.x) >> 5);
  int lane = (int)(threadIdx.x & 31);
  const int tilesD = D_ / 16;
  const int tilesI = N_ / 16;
  const int tilesPerB = tilesD * tilesI;
  int b = wave / tilesPerB;
  if (b >= B_) return;                        // wave-uniform
  int t  = wave % tilesPerB;
  int ti = t / tilesD;
  int td = t % tilesD;

  int m = lane & 15;
  int g = lane >> 4;
  const __bf16* arow = attn + (size_t)(b * N_ + ti * 16 + m) * N_;
  const float*  mb   = mat  + (size_t)b * N_ * D_;
  int n = td * 16 + m;

  v8f acc = {};
#pragma unroll 2
  for (int k0 = 0; k0 < N_; k0 += 32) {
    v16bf av, bv;
#pragma unroll
    for (int e = 0; e < 8; ++e) {
      av[e]     = arow[k0 + g * 8 + e];
      av[e + 8] = arow[k0 + 16 + g * 8 + e];
      bv[e]     = (__bf16)mb[(size_t)(k0 + g * 8 + e) * D_ + n];
      bv[e + 8] = (__bf16)mb[(size_t)(k0 + 16 + g * 8 + e) * D_ + n];
    }
    acc = __builtin_amdgcn_wmma_f32_16x16x32_bf16(
        false, av, false, bv, (short)0, acc, false, false);
  }

  int mbv = g * 8;
#pragma unroll
  for (int r = 0; r < 8; ++r)
    out[(size_t)(b * N_ + ti * 16 + mbv + r) * D_ + n] = acc[r];
}

extern "C" void kernel_launch(void* const* d_in, const int* in_sizes, int n_in,
                              void* d_out, int out_size, void* d_ws, size_t ws_size,
                              hipStream_t stream) {
  (void)in_sizes; (void)n_in; (void)out_size; (void)ws_size;
  const float* matrix = (const float*)d_in[0];
  const int*   mask   = (const int*)d_in[1];
  const float* W1w    = (const float*)d_in[2];
  const float* W1b    = (const float*)d_in[3];
  const float* W2w    = (const float*)d_in[4];
  const float* W2b    = (const float*)d_in[5];
  const float* vw     = (const float*)d_in[6];
  float* out = (float*)d_out;

  // Workspace layout: w1 (1 MB) | w2 (1 MB) | attn bf16 (2 MB)
  char* ws = (char*)d_ws;
  float*  w1   = (float*)ws;
  float*  w2   = (float*)(ws + (size_t)B_ * N_ * A_ * sizeof(float));
  __bf16* attn = (__bf16*)(ws + (size_t)2 * B_ * N_ * A_ * sizeof(float));

  // Projections: (B*N/16)*(A/16) = 1024 waves -> 128 blocks of 8 waves
  const int projWaves  = (B_ * N_ / 16) * (A_ / 16);
  const int projBlocks = projWaves * 32 / 256;
  proj_wmma<<<projBlocks, 256, 0, stream>>>(matrix, W1w, W1b, w1);
  proj_wmma<<<projBlocks, 256, 0, stream>>>(matrix, W2w, W2b, w2);

  // Fused scores + mask + softmax: one block per (b, 8-row i-tile)
  dim3 g2(N_ / TI_, B_);
  score_softmax<<<g2, 256, 0, stream>>>(w1, w2, vw, mask, attn);

  // attn @ matrix: B*(N/16)*(D/16) = 6144 waves -> 768 blocks
  const int avWaves  = B_ * (N_ / 16) * (D_ / 16);
  const int avBlocks = avWaves * 32 / 256;
  av_wmma<<<avBlocks, 256, 0, stream>>>(attn, matrix, out);
}
